// SimplifiedMultiGCN_47047071761043
// MI455X (gfx1250) — compile-verified, matrix operand order checked by reading, not run on hardware
//
#include <hip/hip_runtime.h>
#include <hip/hip_bf16.h>
#include <stddef.h>

typedef __attribute__((ext_vector_type(16))) _Float16 v16h;
typedef __attribute__((ext_vector_type(8)))  _Float16 v8h;
typedef __attribute__((ext_vector_type(8)))  float    v8f;

#define DFEAT 128
#define NODES_PER_GRAPH 500
#define NEG_SLOPE 0.01f
#define BN_EPS 1e-5f

__device__ __forceinline__ float lrelu(float v) {
    return v > 0.0f ? v : NEG_SLOPE * v;
}

// ---------------------------------------------------------------------------
// Degree / normalization
// ---------------------------------------------------------------------------
__global__ void init_deg_kernel(float* deg, int n) {
    int i = blockIdx.x * blockDim.x + threadIdx.x;
    if (i < n) deg[i] = 1.0f;   // self loop
}

__global__ void deg_count_kernel(const int* ei, float* deg, int E) {
    int e = blockIdx.x * blockDim.x + threadIdx.x;
    if (e < E) atomicAdd(&deg[ei[E + e]], 1.0f);   // dst = ei[1][e]
}

__global__ void deg_rsqrt_kernel(float* deg, int n) {
    int i = blockIdx.x * blockDim.x + threadIdx.x;
    if (i < n) deg[i] = rsqrtf(deg[i]);   // in-place -> dinv
}

// ---------------------------------------------------------------------------
// Repack W (128x128 f32, row-major [K][N]) into CDNA5 WMMA B-fragment order:
// Wfrag[((w*4 + chunk)*32 + lane)*16 + j]
//   n  = w*16 + (lane & 15), hi = lane >> 4
//   k  = chunk*32 + (j<8 ? hi*8 + j : 16 + hi*8 + (j-8))
// so each lane loads its v16h fragment as one contiguous 32B global load.
// ---------------------------------------------------------------------------
__global__ void wfrag_kernel(const float* __restrict__ W, _Float16* __restrict__ Wfrag) {
    int tid = blockIdx.x * blockDim.x + threadIdx.x;   // 0 .. 16383
    int j     =  tid        & 15;
    int lane  = (tid >> 4)  & 31;
    int chunk = (tid >> 9)  & 3;
    int w     =  tid >> 11;
    int n  = (w << 4) + (lane & 15);
    int hi =  lane >> 4;
    int k  = chunk * 32 + ((j < 8) ? (hi * 8 + j) : (16 + hi * 8 + (j - 8)));
    Wfrag[tid] = (_Float16)W[k * DFEAT + n];
}

// ---------------------------------------------------------------------------
// Fused GCN dense stage:
//   hW[n,:]   = act(in[n,:]) @ W            (act = lrelu for layers 2,3)
//   outInit   = hW * dinv[n]^2 + bias       (self-loop term, scatter target)
// One block = 16 rows, 8 waves; wave w owns the 16x16 tile of columns
// [w*16, w*16+16). K = 128 -> 4 chained v_wmma_f32_16x16x32_f16.
// A tile staged in LDS in A-fragment order (f32 -> f16, activation fused).
// ---------------------------------------------------------------------------
__global__ void gemm_kernel(const float* __restrict__ in,
                            const _Float16* __restrict__ Wfrag,
                            float* __restrict__ hW,
                            float* __restrict__ outInit,
                            const float* __restrict__ dinv,
                            const float* __restrict__ bias,
                            int n, int applyAct) {
    __shared__ __attribute__((aligned(32))) _Float16 ldsA[4 * 32 * 16]; // 4KB

    const int row0 = blockIdx.x * 16;
    const int tid  = threadIdx.x;        // 0..255
    const int lane = tid & 31;
    const int wave = tid >> 5;           // 0..7  == output column tile

    // --- stage A fragments: thread fills 8 contiguous halves (one b128 store)
    {
        int base  = tid * 8;             // 0..2047 step 8
        int chunk =  base >> 9;
        int flane = (base >> 4) & 31;
        int j0    =  base & 15;          // 0 or 8
        int m     = flane & 15;
        int hi    = flane >> 4;
        int k0    = chunk * 32 + ((j0 == 0) ? (hi * 8) : (16 + hi * 8));
        int row   = row0 + m;
        v8h v;
        #pragma unroll
        for (int jj = 0; jj < 8; ++jj) {
            float x = (row < n) ? in[row * DFEAT + k0 + jj] : 0.0f;
            if (applyAct) x = lrelu(x);
            v[jj] = (_Float16)x;
        }
        *(v8h*)&ldsA[base] = v;
    }
    __syncthreads();

    // --- load fragments
    v16h a0 = *(const v16h*)&ldsA[0 * 512 + lane * 16];
    v16h a1 = *(const v16h*)&ldsA[1 * 512 + lane * 16];
    v16h a2 = *(const v16h*)&ldsA[2 * 512 + lane * 16];
    v16h a3 = *(const v16h*)&ldsA[3 * 512 + lane * 16];

    const _Float16* wb = Wfrag + ((size_t)wave * 4) * 32 * 16;
    v16h b0 = *(const v16h*)&wb[0 * 512 + lane * 16];
    v16h b1 = *(const v16h*)&wb[1 * 512 + lane * 16];
    v16h b2 = *(const v16h*)&wb[2 * 512 + lane * 16];
    v16h b3 = *(const v16h*)&wb[3 * 512 + lane * 16];

    v8f acc = {};
    acc = __builtin_amdgcn_wmma_f32_16x16x32_f16(false, a0, false, b0, (short)0, acc, false, false);
    acc = __builtin_amdgcn_wmma_f32_16x16x32_f16(false, a1, false, b1, (short)0, acc, false, false);
    acc = __builtin_amdgcn_wmma_f32_16x16x32_f16(false, a2, false, b2, (short)0, acc, false, false);
    acc = __builtin_amdgcn_wmma_f32_16x16x32_f16(false, a3, false, b3, (short)0, acc, false, false);

    // --- store 16x16 f32 tile; VGPR r holds M = r + (lane>>4)*8, N = lane&15
    const int mbase = row0 + ((lane >> 4) << 3);
    const int nn    = wave * 16 + (lane & 15);
    const float bn  = bias[nn];
    if (row0 + 16 <= n) {       // uniform fast path: full tile, no guards
        #pragma unroll
        for (int r = 0; r < 8; ++r) {
            int row  = mbase + r;
            float di = dinv[row];
            hW[row * DFEAT + nn]      = acc[r];
            outInit[row * DFEAT + nn] = acc[r] * di * di + bn;
        }
    } else {
        #pragma unroll
        for (int r = 0; r < 8; ++r) {
            int row = mbase + r;
            if (row < n) {
                float di = dinv[row];
                hW[row * DFEAT + nn]      = acc[r];
                outInit[row * DFEAT + nn] = acc[r] * di * di + bn;
            }
        }
    }
}

// ---------------------------------------------------------------------------
// Edge scatter: out[dst] += hW[src] * dinv[src]*dinv[dst]
// One 32-lane group per edge; each lane moves a float4 (128 feats total).
// hW (25.6 MB) stays L2-resident; atomics resolve in L2.
// ---------------------------------------------------------------------------
__global__ void edge_kernel(const float* __restrict__ hW, const int* __restrict__ ei,
                            const float* __restrict__ dinv, float* __restrict__ out, int E) {
    int e = blockIdx.x * 8 + (threadIdx.x >> 5);
    if (e >= E) return;
    int lane = threadIdx.x & 31;
    int src = ei[e];
    int dst = ei[E + e];
    float nrm = dinv[src] * dinv[dst];
    const float4 v = *(const float4*)(hW + src * DFEAT + lane * 4);
    float* o = out + dst * DFEAT + lane * 4;
    atomicAdd(o + 0, v.x * nrm);
    atomicAdd(o + 1, v.y * nrm);
    atomicAdd(o + 2, v.z * nrm);
    atomicAdd(o + 3, v.w * nrm);
}

// ---------------------------------------------------------------------------
// Global mean pool (batches contiguous, 500 nodes/graph); final-layer
// activation fused on read -> f[g][0..127]
// ---------------------------------------------------------------------------
__global__ void pool_kernel(const float* __restrict__ h, float* __restrict__ f) {
    int g = blockIdx.x;
    int d = threadIdx.x;          // 128 threads
    const float* base = h + (size_t)g * NODES_PER_GRAPH * DFEAT + d;
    float s = 0.0f;
    for (int i = 0; i < NODES_PER_GRAPH; ++i) s += lrelu(base[i * DFEAT]);
    f[g * 138 + d] = s * (1.0f / NODES_PER_GRAPH);
}

// ---------------------------------------------------------------------------
// ogt embedding MLP 1 -> 20 -> 10 -> f[g][128..137]
// ---------------------------------------------------------------------------
__global__ void ogt_kernel(const float* __restrict__ ogt,
                           const float* __restrict__ Wo1, const float* __restrict__ bo1,
                           const float* __restrict__ Wo2, const float* __restrict__ bo2,
                           float* __restrict__ f, int G) {
    int g = threadIdx.x;
    if (g >= G) return;
    float t = ogt[g];
    float o1[20];
    #pragma unroll
    for (int j = 0; j < 20; ++j) o1[j] = lrelu(t * Wo1[j] + bo1[j]);
    #pragma unroll
    for (int k = 0; k < 10; ++k) {
        float acc = bo2[k];
        #pragma unroll
        for (int j = 0; j < 20; ++j) acc += o1[j] * Wo2[j * 10 + k];
        f[g * 138 + 128 + k] = lrelu(acc);
    }
}

// ---------------------------------------------------------------------------
// FC head
// ---------------------------------------------------------------------------
__global__ void fc1_kernel(const float* __restrict__ f, const float* __restrict__ Wf1,
                           const float* __restrict__ bf1, float* __restrict__ f1) {
    __shared__ float sf[138];
    int g = blockIdx.x;
    for (int k = threadIdx.x; k < 138; k += blockDim.x) sf[k] = f[g * 138 + k];
    __syncthreads();
    int j = threadIdx.x;
    if (j >= 92) return;
    float acc = bf1[j];
    for (int k = 0; k < 138; ++k) acc += sf[k] * Wf1[k * 92 + j];
    f1[g * 92 + j] = lrelu(acc);
}

__global__ void stats_kernel(const float* __restrict__ v, int rows, int cols,
                             float* __restrict__ m, float* __restrict__ inv) {
    int j = threadIdx.x;
    if (j >= cols) return;
    float s = 0.0f;
    for (int g = 0; g < rows; ++g) s += v[g * cols + j];
    float mean = s / rows;
    float vs = 0.0f;
    for (int g = 0; g < rows; ++g) { float d = v[g * cols + j] - mean; vs += d * d; }
    m[j]   = mean;
    inv[j] = rsqrtf(vs / rows + BN_EPS);
}

__global__ void fc2_kernel(const float* __restrict__ f1, const float* __restrict__ m1,
                           const float* __restrict__ iv1, const float* __restrict__ g1,
                           const float* __restrict__ be1, const float* __restrict__ Wf2,
                           const float* __restrict__ bf2, float* __restrict__ f2) {
    __shared__ float sn[92];
    int g = blockIdx.x;
    for (int j = threadIdx.x; j < 92; j += blockDim.x)
        sn[j] = (f1[g * 92 + j] - m1[j]) * iv1[j] * g1[j] + be1[j];
    __syncthreads();
    int k = threadIdx.x;
    if (k >= 46) return;
    float acc = bf2[k];
    for (int j = 0; j < 92; ++j) acc += sn[j] * Wf2[j * 46 + k];
    f2[g * 46 + k] = lrelu(acc);
}

__global__ void fc3_kernel(const float* __restrict__ f2, const float* __restrict__ m2,
                           const float* __restrict__ iv2, const float* __restrict__ g2,
                           const float* __restrict__ be2, const float* __restrict__ Wf3,
                           const float* __restrict__ bf3, float* __restrict__ out, int G) {
    int g = threadIdx.x;
    if (g >= G) return;
    float acc = bf3[0];
    for (int k = 0; k < 46; ++k)
        acc += ((f2[g * 46 + k] - m2[k]) * iv2[k] * g2[k] + be2[k]) * Wf3[k];
    out[g] = acc;
}

// ---------------------------------------------------------------------------
extern "C" void kernel_launch(void* const* d_in, const int* in_sizes, int n_in,
                              void* d_out, int out_size, void* d_ws, size_t ws_size,
                              hipStream_t stream) {
    const float* x    = (const float*)d_in[0];
    const int*   ei   = (const int*)  d_in[1];
    const float* ogt  = (const float*)d_in[3];
    const float* W[3] = {(const float*)d_in[4], (const float*)d_in[6], (const float*)d_in[8]};
    const float* b[3] = {(const float*)d_in[5], (const float*)d_in[7], (const float*)d_in[9]};
    const float* Wo1  = (const float*)d_in[10];
    const float* bo1  = (const float*)d_in[11];
    const float* Wo2  = (const float*)d_in[12];
    const float* bo2  = (const float*)d_in[13];
    const float* Wf1  = (const float*)d_in[14];
    const float* bf1  = (const float*)d_in[15];
    const float* g1   = (const float*)d_in[16];
    const float* be1  = (const float*)d_in[17];
    const float* Wf2  = (const float*)d_in[18];
    const float* bf2  = (const float*)d_in[19];
    const float* g2   = (const float*)d_in[20];
    const float* be2  = (const float*)d_in[21];
    const float* Wf3  = (const float*)d_in[22];
    const float* bf3  = (const float*)d_in[23];
    float* out = (float*)d_out;

    const int N = in_sizes[0] / DFEAT;   // 50000
    const int E = in_sizes[1] / 2;       // 800000
    const int G = in_sizes[3];           // 100
    const int NF = N * DFEAT;

    // workspace carve-out (256B aligned blocks)
    char* p = (char*)d_ws;
    auto alloc = [&](size_t bytes) { void* r = (void*)p; p += (bytes + 255) & ~(size_t)255; return r; };
    float*     dinv  = (float*)    alloc((size_t)N * 4);
    float*     P0    = (float*)    alloc((size_t)NF * 4);   // hW scratch
    float*     P1    = (float*)    alloc((size_t)NF * 4);
    float*     P2    = (float*)    alloc((size_t)NF * 4);
    _Float16*  Wfrag = (_Float16*) alloc(16384 * 2);
    float*     fcat  = (float*)    alloc((size_t)G * 138 * 4);
    float*     f1    = (float*)    alloc((size_t)G * 92 * 4);
    float*     m1    = (float*)    alloc(92 * 4);
    float*     iv1   = (float*)    alloc(92 * 4);
    float*     f2    = (float*)    alloc((size_t)G * 46 * 4);
    float*     m2    = (float*)    alloc(46 * 4);
    float*     iv2   = (float*)    alloc(46 * 4);

    // degree -> dinv
    init_deg_kernel <<<(N + 255) / 256, 256, 0, stream>>>(dinv, N);
    deg_count_kernel<<<(E + 255) / 256, 256, 0, stream>>>(ei, dinv, E);
    deg_rsqrt_kernel<<<(N + 255) / 256, 256, 0, stream>>>(dinv, N);

    // 3 GCN layers; ping-pong: x -> P1 -> P2 -> P1
    // Activation of each layer is applied lazily on the next read.
    const float* hin     = x;
    float*       hout[3] = {P1, P2, P1};
    for (int l = 0; l < 3; ++l) {
        wfrag_kernel<<<64, 256, 0, stream>>>(W[l], Wfrag);
        gemm_kernel <<<(N + 15) / 16, 256, 0, stream>>>(hin, Wfrag, P0, hout[l],
                                                        dinv, b[l], N, l > 0 ? 1 : 0);
        edge_kernel <<<(E + 7) / 8, 256, 0, stream>>>(P0, ei, dinv, hout[l], E);
        hin = hout[l];
    }

    // pooling (activation fused) + ogt embedding -> concat feature [G,138]
    pool_kernel<<<G, DFEAT, 0, stream>>>(hout[2], fcat);
    ogt_kernel <<<1, 128, 0, stream>>>(ogt, Wo1, bo1, Wo2, bo2, fcat, G);

    // FC head with batchnorm
    fc1_kernel  <<<G, 96, 0, stream>>>(fcat, Wf1, bf1, f1);
    stats_kernel<<<1, 96, 0, stream>>>(f1, G, 92, m1, iv1);
    fc2_kernel  <<<G, 64, 0, stream>>>(f1, m1, iv1, g1, be1, Wf2, bf2, f2);
    stats_kernel<<<1, 64, 0, stream>>>(f2, G, 46, m2, iv2);
    fc3_kernel  <<<1, 128, 0, stream>>>(f2, m2, iv2, g2, be2, Wf3, bf3, out, G);
}
